// RNNAutoencoder_74947179315227
// MI455X (gfx1250) — compile-verified, matrix operand order checked by reading, not run on hardware
//
#include <hip/hip_runtime.h>

// ---------------------------------------------------------------------------
// GRU autoencoder, persistent WMMA kernel for gfx1250 (MI455X).
// B=128, S=1024, D=H=256, gates G=3*256=768.
//
// Grid: 16 blocks (hTile = 16-col slice of H) x 256 threads (8 waves = mTile,
// 16-row slice of B). Each wave owns a 16x16 tile of the hidden state and
// computes 6 WMMA accumulators per step (r/z/n for gi and gh).
// Weights live in LDS in bf16; hidden state: bf16 double-buffered in global ws
// (for cross-wave WMMA A-loads) + f32 carried in registers (each wave's own
// tile, layout-invariant across steps). One software grid barrier per step.
// ---------------------------------------------------------------------------

typedef __attribute__((ext_vector_type(16))) __bf16 v16bf;
typedef __attribute__((ext_vector_type(8)))  __bf16 v8bf;
typedef __attribute__((ext_vector_type(8)))  float  v8f;

#define B_   128
#define S_   1024
#define D_   256
#define H_   256
#define KSTEPS 8            // 256 / 32
#define LPAD 272            // padded LDS row stride (elements): 544B -> 2-way conflicts
#define NBLK 16             // grid size (hTiles)

// ---------------------------- x -> bf16 convert ----------------------------
__global__ void xcvt_kernel(const float* __restrict__ x, __bf16* __restrict__ xb) {
  long i = ((long)blockIdx.x * blockDim.x + threadIdx.x) * 8;
  float4 a = *(const float4*)(x + i);
  float4 b = *(const float4*)(x + i + 4);
  v8bf o;
  o[0] = (__bf16)a.x; o[1] = (__bf16)a.y; o[2] = (__bf16)a.z; o[3] = (__bf16)a.w;
  o[4] = (__bf16)b.x; o[5] = (__bf16)b.y; o[6] = (__bf16)b.z; o[7] = (__bf16)b.w;
  *(v8bf*)(xb + i) = o;
}

// ---------------------------- helpers --------------------------------------
__device__ __forceinline__ float sigmoidf_(float x) {
  return 1.0f / (1.0f + __expf(-x));
}

// Load this block's 6 weight column-tiles (3 gates x {W_ih, W_hh}) into LDS.
// Both matrices are [768, 256] row-major f32; gate g rows are [g*256, g*256+256).
__device__ __forceinline__ void load_w(__bf16* lds, const float* __restrict__ Wih,
                                       const float* __restrict__ Whh, int hTile) {
  for (int idx = threadIdx.x; idx < 6 * 16 * 256; idx += 256) {
    int tile = idx >> 12;       // 0..5
    int rem  = idx & 4095;
    int n    = rem >> 8;        // 0..15 (output column within tile)
    int k    = rem & 255;       // 0..255
    const float* W = (tile < 3) ? Wih : Whh;
    int g = (tile < 3) ? tile : tile - 3;
    lds[(tile * 16 + n) * LPAD + k] =
        (__bf16)W[(g * 256 + (hTile << 4) + n) * 256 + k];
  }
}

// One GEMM triple: C{r,z,n} += A[16x256] * W{r,z,n}[256x16], K swept 32 at a time.
// A fragment (v16bf): lane l holds row l%16, contiguous K chunk at 16*(l/16).
// B fragment: lane l holds weight row (= output col) l%16, same K chunk — i.e.
// the row-major weight layout directly. A regs reused across the 3 gates.
__device__ __forceinline__ void gemm3(const __bf16* __restrict__ A, int rowStride,
                                      const __bf16* __restrict__ W, int lane,
                                      v8f& cr, v8f& cz, v8f& cn) {
  const int m  = lane & 15;
  const int kh = (lane >> 4) << 4;   // 0 or 16
  const __bf16* ap = A + (long)m * rowStride + kh;
  const __bf16* wr = W + m * LPAD + kh;
  const __bf16* wz = wr + 16 * LPAD;
  const __bf16* wn = wz + 16 * LPAD;
#pragma unroll
  for (int kt = 0; kt < KSTEPS; ++kt) {
    v16bf a  = *(const v16bf*)(ap + kt * 32);
    v16bf b0 = *(const v16bf*)(wr + kt * 32);
    v16bf b1 = *(const v16bf*)(wz + kt * 32);
    v16bf b2 = *(const v16bf*)(wn + kt * 32);
    cr = __builtin_amdgcn_wmma_f32_16x16x32_bf16(false, a, false, b0, (short)0, cr, false, false);
    cz = __builtin_amdgcn_wmma_f32_16x16x32_bf16(false, a, false, b1, (short)0, cz, false, false);
    cn = __builtin_amdgcn_wmma_f32_16x16x32_bf16(false, a, false, b2, (short)0, cn, false, false);
  }
}

// Gate math + hidden-state update. C layout: lane l, vgpr j -> row j+8*(l>>4),
// col l&15 — identical for all six accumulators AND step-invariant, so the
// wave's own f32 h tile is carried in `hprev` registers across all steps.
__device__ __forceinline__ void finish_step(
    const v8f& ir, const v8f& iz, const v8f& in_,
    const v8f& hr, const v8f& hz, const v8f& hn,
    float br, float bz, float bi, float bh,
    v8f& hprev, __bf16* __restrict__ hb_wr, float* __restrict__ outp,
    int lane, int mTile, int hTile, int t) {
  const int col    = (hTile << 4) + (lane & 15);
  const int rowTop = (mTile << 4) + ((lane >> 4) << 3);
#pragma unroll
  for (int j = 0; j < 8; ++j) {
    const int row = rowTop + j;
    float r  = sigmoidf_(ir[j] + hr[j] + br);
    float z  = sigmoidf_(iz[j] + hz[j] + bz);
    float nn = tanhf(in_[j] + bi + r * (hn[j] + bh));
    float hnew = (1.0f - z) * nn + z * hprev[j];
    hprev[j] = hnew;
    hb_wr[row * H_ + col] = (__bf16)hnew;
    if (outp) outp[(long)row * (S_ * D_) + (long)t * D_ + col] = hnew;
  }
}

// Software grid barrier: monotonically increasing counter, 16 resident blocks.
__device__ __forceinline__ void grid_barrier(unsigned* cnt, unsigned target) {
  __syncthreads();
  __threadfence();                       // make this block's stores agent-visible
  if (threadIdx.x == 0) {
    __hip_atomic_fetch_add(cnt, 1u, __ATOMIC_RELEASE, __HIP_MEMORY_SCOPE_AGENT);
    while (__hip_atomic_load(cnt, __ATOMIC_ACQUIRE, __HIP_MEMORY_SCOPE_AGENT) < target)
      __builtin_amdgcn_s_sleep(2);
  }
  __syncthreads();
}

// ---------------------------- persistent kernel ----------------------------
__global__ void __launch_bounds__(256, 1)
rnn_kernel(const __bf16* __restrict__ xb, const int* __restrict__ tf_mask,
           const float* __restrict__ eWih, const float* __restrict__ eWhh,
           const float* __restrict__ eBih, const float* __restrict__ eBhh,
           const float* __restrict__ dWih, const float* __restrict__ dWhh,
           const float* __restrict__ dBih, const float* __restrict__ dBhh,
           __bf16* __restrict__ hb, float* __restrict__ out,
           unsigned* __restrict__ cnt) {
  __shared__ __bf16 ldsW[6 * 16 * LPAD];   // 52224 B

  const int hTile = blockIdx.x;            // 0..15
  const int mTile = threadIdx.x >> 5;      // 0..7
  const int lane  = threadIdx.x & 31;
  const int m     = lane & 15;
  const int kh    = (lane >> 4) << 4;

  // Stage encoder weights; zero ping-buffer 0 of the bf16 hidden state (h0=0).
  load_w(ldsW, eWih, eWhh, hTile);
  {
    const int col    = (hTile << 4) + m;
    const int rowTop = (mTile << 4) + ((lane >> 4) << 3);
#pragma unroll
    for (int j = 0; j < 8; ++j) hb[(rowTop + j) * H_ + col] = (__bf16)0.0f;
  }
  unsigned bar = 1;
  grid_barrier(cnt, NBLK * bar);

  const int col = (hTile << 4) + m;
  const long xRowStride = (long)S_ * D_;
  const __bf16* xbW = xb + (long)(mTile << 4) * xRowStride;  // this wave's batch rows
  const __bf16* xbL = xbW + (long)m * xRowStride + kh;       // lane's x A-load base
  const v8f vzero = {0.f, 0.f, 0.f, 0.f, 0.f, 0.f, 0.f, 0.f};
  v8f hprev = vzero;                                          // registered f32 h tile

  // ---------------- encoder: h_{t+1} = GRU(x_t, h_t) ----------------
  float br = eBih[col] + eBhh[col];
  float bz = eBih[H_ + col] + eBhh[H_ + col];
  float bi = eBih[2 * H_ + col];
  float bh = eBhh[2 * H_ + col];

  int ping = 0;
  for (int t = 0; t < S_; ++t) {
    const __bf16* hb_rd = hb + ping * (B_ * H_);
    __bf16*       hb_wr = hb + (ping ^ 1) * (B_ * H_);

    if (t + 1 < S_) __builtin_prefetch(xbL + (long)(t + 1) * D_, 0, 1);

    v8f ir = vzero, iz = vzero, in_ = vzero, hr = vzero, hz = vzero, hn = vzero;
    gemm3(xbW + (long)t * D_, (int)xRowStride, ldsW, lane, ir, iz, in_);
    gemm3(hb_rd + (mTile << 4) * H_, H_, ldsW + 3 * 16 * LPAD, lane, hr, hz, hn);
    finish_step(ir, iz, in_, hr, hz, hn, br, bz, bi, bh,
                hprev, hb_wr, nullptr, lane, mTile, hTile, t);
    ping ^= 1;
    ++bar;
    grid_barrier(cnt, NBLK * bar);
  }

  // ---------------- decoder: teacher forcing ----------------
  load_w(ldsW, dWih, dWhh, hTile);   // safe: grid barrier ended with __syncthreads
  __syncthreads();
  br = dBih[col] + dBhh[col];
  bz = dBih[H_ + col] + dBhh[H_ + col];
  bi = dBih[2 * H_ + col];
  bh = dBhh[2 * H_ + col];

  for (int t = 0; t < S_; ++t) {
    const __bf16* hb_rd = hb + ping * (B_ * H_);
    __bf16*       hb_wr = hb + (ping ^ 1) * (B_ * H_);

    // Possible next-step input is x_t (if tf_mask[t]): prefetch its tile.
    __builtin_prefetch(xbL + (long)t * D_, 0, 1);

    // d_in(t): x[:,0] at t=0; else tf_mask[t-1] ? x[:,t-1] : h_new(t-1)(=hb_rd)
    const __bf16* Ain; int As;
    if (t == 0)                 { Ain = xbW;                         As = (int)xRowStride; }
    else if (tf_mask[t - 1])    { Ain = xbW + (long)(t - 1) * D_;    As = (int)xRowStride; }
    else                        { Ain = hb_rd + (mTile << 4) * H_;   As = H_; }

    v8f ir = vzero, iz = vzero, in_ = vzero, hr = vzero, hz = vzero, hn = vzero;
    gemm3(Ain, As, ldsW, lane, ir, iz, in_);
    gemm3(hb_rd + (mTile << 4) * H_, H_, ldsW + 3 * 16 * LPAD, lane, hr, hz, hn);
    finish_step(ir, iz, in_, hr, hz, hn, br, bz, bi, bh,
                hprev, hb_wr, out, lane, mTile, hTile, t);
    ping ^= 1;
    ++bar;
    grid_barrier(cnt, NBLK * bar);
  }
}

// ---------------------------- launcher --------------------------------------
extern "C" void kernel_launch(void* const* d_in, const int* in_sizes, int n_in,
                              void* d_out, int out_size, void* d_ws, size_t ws_size,
                              hipStream_t stream) {
  (void)in_sizes; (void)n_in; (void)out_size; (void)ws_size;
  const float* x     = (const float*)d_in[0];
  const int*   tf    = (const int*)d_in[1];     // tf_mask (bool -> int per harness)
  const float* eWih  = (const float*)d_in[2];
  const float* eWhh  = (const float*)d_in[3];
  const float* eBih  = (const float*)d_in[4];
  const float* eBhh  = (const float*)d_in[5];
  const float* dWih  = (const float*)d_in[6];
  const float* dWhh  = (const float*)d_in[7];
  const float* dBih  = (const float*)d_in[8];
  const float* dBhh  = (const float*)d_in[9];
  float* out = (float*)d_out;

  // Workspace layout (all 32B-aligned):
  //   [0,256)                : grid-barrier counter
  //   [256, 256+64MB)        : x in bf16
  //   next 128KB             : hb (2 ping-pong bf16 hidden buffers, 2*B*H)
  char* ws = (char*)d_ws;
  unsigned* cnt  = (unsigned*)ws;
  __bf16*   xb   = (__bf16*)(ws + 256);
  size_t xbBytes = (size_t)B_ * S_ * D_ * 2;                 // 64 MB
  __bf16*   hbuf = (__bf16*)(ws + 256 + xbBytes);            // 2*B*H bf16

  hipMemsetAsync(ws, 0, 256, stream);  // zero barrier counter (capture-legal)

  const long nx = (long)B_ * S_ * D_;  // 33,554,432 (divisible by 256*8)
  xcvt_kernel<<<(unsigned)(nx / (256 * 8)), 256, 0, stream>>>(x, xb);

  rnn_kernel<<<NBLK, 256, 0, stream>>>(xb, tf, eWih, eWhh, eBih, eBhh,
                                       dWih, dWhh, dBih, dBhh,
                                       hbuf, out, cnt);
}
